// Attention_org_80771154968910
// MI455X (gfx1250) — matrix operand order repaired and checked by reading, not compiled, not run on hardware
//
#include <hip/hip_runtime.h>

// ---------------------------------------------------------------------------
// CDNA5 (gfx1250) multi-branch attention, WMMA f16 / f32-accumulate pipeline.
//   1) K,V projections (fp32 in -> f16 out, K stored transposed)
//   2) per-branch: q proj -> scores GEMM (+fused instance-norm stats via
//      atomics) -> norm+softmax -> ctx GEMM (2-head accumulate) -> out proj
// One templated, double-buffered 64x64-tile WMMA GEMM covers every matmul:
//   - f16 operands stream into LDS via GLOBAL_LOAD_ASYNC_TO_LDS_B128
//     (ASYNCcnt), overlapped with the WMMA block of the previous tile.
//   - f32 operands use register-prefetch pipelining (global->VGPR before
//     compute, cvt+ds_store after), so loads overlap the matrix math too.
// ---------------------------------------------------------------------------

typedef _Float16 h16;
typedef __attribute__((ext_vector_type(16))) _Float16 v16h;
typedef __attribute__((ext_vector_type(8)))  _Float16 v8h;
typedef __attribute__((ext_vector_type(8)))  float    v8f;
typedef __attribute__((ext_vector_type(4)))  float    v4f;

#define LDT 40  // LDS row stride in halves: 32 + 8 pad; 80B -> 16B aligned rows

struct GP {
  const void* A; const void* B; void* O;
  long aS0, aS1, bS0, bS1, oS0, oS1;   // z = z1*z0n + z0 offset strides (elems)
  long aSrc, bSrc;                     // multi-source (per-head) strides
  int  z0n, nsrc;
  int  lda, ldb, ldo, K;               // leading dims (elems), contraction len
  float alpha;
  int  flags;                          // bit0: f16 output, bit1: transposed out
  float* stats;                        // per-z {sum,sumsq} atomic accumulators
};

// ---- two-phase tile movers: 64 rows x 32 K, K-contiguous global & LDS ------
// f32 source: fetch() pulls global->VGPRs, commit() cvt+stores into LDS.
struct FetchF32 { v4f r[4]; };
__device__ __forceinline__ void tile_fetch(FetchF32& st, h16*, const float* src, int ld) {
  int t = threadIdx.x;
#pragma unroll
  for (int i = 0; i < 4; ++i) {
    int idx = t + i * 128;            // 0..511 float4s
    int r = idx >> 3, c4 = idx & 7;
    st.r[i] = *(const v4f*)(src + (long)r * ld + c4 * 4);
  }
}
__device__ __forceinline__ void tile_commit(FetchF32& st, h16* lds) {
  int t = threadIdx.x;
#pragma unroll
  for (int i = 0; i < 4; ++i) {
    int idx = t + i * 128;
    int r = idx >> 3, c4 = idx & 7;
    h16* d = lds + r * LDT + c4 * 4;
    d[0] = (h16)st.r[i][0]; d[1] = (h16)st.r[i][1];
    d[2] = (h16)st.r[i][2]; d[3] = (h16)st.r[i][3];
  }
}
// f16 source: fetch() issues per-lane async DMA global->LDS (ASYNCcnt),
// 16B per op; commit() is a no-op. Drained by s_wait_asynccnt after compute.
struct FetchF16 {};
__device__ __forceinline__ void tile_fetch(FetchF16&, h16* lds, const h16* src, int ld) {
  int t = threadIdx.x;
#pragma unroll
  for (int i = 0; i < 2; ++i) {
    int idx = t + i * 128;            // 0..255 16B chunks
    int r = idx >> 2, c8 = idx & 3;
    unsigned long long ga = (unsigned long long)(const void*)(src + (long)r * ld + c8 * 8);
    unsigned la = (unsigned)(unsigned long long)(const void*)(lds + r * LDT + c8 * 8);
    asm volatile("global_load_async_to_lds_b128 %0, %1, off"
                 :: "v"(la), "v"(ga) : "memory");
  }
}
__device__ __forceinline__ void tile_commit(FetchF16&, h16*) {}

template <typename T> struct FetchOf;
template <> struct FetchOf<float> { using type = FetchF32; };
template <> struct FetchOf<h16>   { using type = FetchF16; };

// ---- fragment gathers matching ISA 16-bit WMMA VGPR layouts ----
// A 16x32: lane(0-15) row=lane, halves0-7:K=kb..kb+7, halves8-15:K=kb+16..+23
//          (kb = 0 for lanes 0-15, 8 for lanes 16-31)
__device__ __forceinline__ v16h frag_a(const h16* lds, int row0) {
  int lane = threadIdx.x & 31;
  const h16* p = lds + (row0 + (lane & 15)) * LDT;
  int kb = (lane & 16) ? 8 : 0;
  v8h lo = *(const v8h*)(p + kb);
  v8h hi = *(const v8h*)(p + kb + 16);
  return __builtin_shufflevector(lo, hi, 0,1,2,3,4,5,6,7,8,9,10,11,12,13,14,15);
}
// B 32x16: lane col N=lane&15, halves0-7:K=kb..kb+7, halves8-15:K=kb+16..+23 (kb = 0 / 8)
__device__ __forceinline__ v16h frag_b(const h16* lds, int col0) {
  int lane = threadIdx.x & 31;
  const h16* p = lds + (col0 + (lane & 15)) * LDT;
  int kb = (lane & 16) ? 8 : 0;
  v8h lo = *(const v8h*)(p + kb);
  v8h hi = *(const v8h*)(p + kb + 16);
  return __builtin_shufflevector(lo, hi, 0,1,2,3,4,5,6,7,8,9,10,11,12,13,14,15);
}

template <typename TA, typename TB>
__global__ __launch_bounds__(128)
void gemm_wmma(GP p) {
  __shared__ __align__(16) h16 lA[2][64 * LDT];
  __shared__ __align__(16) h16 lB[2][64 * LDT];
  __shared__ float rs0[128];
  __shared__ float rs1[128];

  int z  = blockIdx.z;
  int z0 = z % p.z0n, z1 = z / p.z0n;
  const TA* Ab = (const TA*)p.A + (long)z1 * p.aS1 + (long)z0 * p.aS0;
  const TB* Bb = (const TB*)p.B + (long)z1 * p.bS1 + (long)z0 * p.bS0;
  int m0 = blockIdx.y * 64, n0 = blockIdx.x * 64;
  int w  = threadIdx.x >> 5;
  int wr = (w >> 1) * 32, wc = (w & 1) * 32;
  int kIt = p.K >> 5;
  int nIter = p.nsrc * kIt;

  typename FetchOf<TA>::type fa;
  typename FetchOf<TB>::type fb;
  auto srcA = [&](int it) { int s = it / kIt, k = (it % kIt) << 5;
                            return Ab + (long)s * p.aSrc + (long)m0 * p.lda + k; };
  auto srcB = [&](int it) { int s = it / kIt, k = (it % kIt) << 5;
                            return Bb + (long)s * p.bSrc + (long)n0 * p.ldb + k; };

  // prologue: fill buffer 0
  tile_fetch(fa, lA[0], srcA(0), p.lda);
  tile_fetch(fb, lB[0], srcB(0), p.ldb);
  tile_commit(fa, lA[0]);
  tile_commit(fb, lB[0]);
  asm volatile("s_wait_asynccnt 0x0" ::: "memory");
  __syncthreads();

  v8f acc[2][2] = {};
  for (int it = 0; it < nIter; ++it) {
    int cur = it & 1, nxt = cur ^ 1;
    if (it + 1 < nIter) {           // start next tile before computing current
      tile_fetch(fa, lA[nxt], srcA(it + 1), p.lda);
      tile_fetch(fb, lB[nxt], srcB(it + 1), p.ldb);
    }
#pragma unroll
    for (int i = 0; i < 2; ++i) {
      v16h a = frag_a(lA[cur], wr + i * 16);
#pragma unroll
      for (int j = 0; j < 2; ++j) {
        v16h b = frag_b(lB[cur], wc + j * 16);
        acc[i][j] = __builtin_amdgcn_wmma_f32_16x16x32_f16(
            false, a, false, b, (short)0, acc[i][j], false, false);
      }
    }
    if (it + 1 < nIter) {           // f32 path: cvt+ds_store after the math
      tile_commit(fa, lA[nxt]);
      tile_commit(fb, lB[nxt]);
    }
    asm volatile("s_wait_asynccnt 0x0" ::: "memory");
    __syncthreads();
  }

  // ---- epilogue: D layout: VGPR r -> M = r + (lane&16 ? 8:0), N = lane&15 ----
  int lane = threadIdx.x & 31;
  int rofs = (lane & 16) ? 8 : 0;
  int colo = lane & 15;
  long zo = (long)z1 * p.oS1 + (long)z0 * p.oS0;
  float sum = 0.f, ssq = 0.f;
#pragma unroll
  for (int i = 0; i < 2; ++i)
#pragma unroll
    for (int j = 0; j < 2; ++j)
#pragma unroll
      for (int r = 0; r < 8; ++r) {
        float v = acc[i][j][r] * p.alpha;
        int gm = m0 + wr + i * 16 + rofs + r;
        int gn = n0 + wc + j * 16 + colo;
        sum += v; ssq += v * v;
        long idx = (p.flags & 2) ? ((long)gn * p.ldo + gm)
                                 : ((long)gm * p.ldo + gn);
        if (p.flags & 1) ((h16*)p.O)[zo + idx] = (h16)v;
        else             ((float*)p.O)[zo + idx] = v;
      }

  if (p.stats) {  // fused instance-norm statistics (scores stage only)
    rs0[threadIdx.x] = sum; rs1[threadIdx.x] = ssq;
    __syncthreads();
    for (int s = 64; s > 0; s >>= 1) {
      if ((int)threadIdx.x < s) {
        rs0[threadIdx.x] += rs0[threadIdx.x + s];
        rs1[threadIdx.x] += rs1[threadIdx.x + s];
      }
      __syncthreads();
    }
    if (threadIdx.x == 0) {
      atomicAdd(&p.stats[2 * z + 0], rs0[0]);
      atomicAdd(&p.stats[2 * z + 1], rs1[0]);
    }
  }
}

__global__ void zero_stats_kernel(float* s) {
  if (threadIdx.x < 64) s[threadIdx.x] = 0.f;
}

// instance-norm (per b,h over C*KV) + softmax along KV; one block per row d
__global__ __launch_bounds__(256)
void norm_softmax_kernel(const float* __restrict__ scores,
                         h16* __restrict__ probs,
                         const float* __restrict__ stats, int C) {
  const int KVv = 960;
  int z = blockIdx.y, d = blockIdx.x, t = threadIdx.x;
  const float* row = scores + ((long)z * C + d) * KVv;
  h16* prow = probs + ((long)z * C + d) * KVv;
  __shared__ float buf[960];
  __shared__ float red[256];
  float cnt = (float)C * 960.0f;
  float mu  = stats[2 * z] / cnt;
  float var = stats[2 * z + 1] / cnt - mu * mu;
  float rstd = rsqrtf(fmaxf(var, 0.f) + 1e-5f);

  float lmax = -3.4e38f;
  for (int i = t; i < KVv; i += 256) {
    float v = (row[i] - mu) * rstd;
    buf[i] = v;
    lmax = fmaxf(lmax, v);
  }
  red[t] = lmax; __syncthreads();
  for (int s = 128; s > 0; s >>= 1) {
    if (t < s) red[t] = fmaxf(red[t], red[t + s]);
    __syncthreads();
  }
  float gmax = red[0]; __syncthreads();
  float lsum = 0.f;
  for (int i = t; i < KVv; i += 256) {
    float e = __expf(buf[i] - gmax);
    buf[i] = e; lsum += e;
  }
  red[t] = lsum; __syncthreads();
  for (int s = 128; s > 0; s >>= 1) {
    if (t < s) red[t] += red[t + s];
    __syncthreads();
  }
  float inv = 1.0f / red[0];
  for (int i = t; i < KVv; i += 256) prow[i] = (h16)(buf[i] * inv);
}

extern "C" void kernel_launch(void* const* d_in, const int* in_sizes, int n_in,
                              void* d_out, int out_size, void* d_ws, size_t ws_size,
                              hipStream_t stream) {
  const int Bb = 16, Nn = 1024, KVv = 960;
  const int Cs[4]  = {64, 128, 256, 512};
  const int off[4] = {0, 64, 192, 448};
  const float SCALE = 0.0322748612f;  // 1/sqrt(960)

  const float* emb[4] = {(const float*)d_in[0], (const float*)d_in[1],
                         (const float*)d_in[2], (const float*)d_in[3]};
  const float* emb_all = (const float*)d_in[4];
  const float* Wq[4] = {(const float*)d_in[5], (const float*)d_in[7],
                        (const float*)d_in[9], (const float*)d_in[11]};
  const float* Wo[4] = {(const float*)d_in[6], (const float*)d_in[8],
                        (const float*)d_in[10], (const float*)d_in[12]};
  const float* Wk = (const float*)d_in[13];
  const float* Wv = (const float*)d_in[14];

  // ---- workspace layout (with overlays) ----
  char* ws = (char*)d_ws;
  size_t o = 0;
  h16* kf = (h16*)(ws + o); o += (size_t)Bb * 2 * KVv * Nn * 2;  // K^T [b,h,m,n]
  h16* vf = (h16*)(ws + o); o += (size_t)Bb * 2 * Nn * KVv * 2;  // V   [b,h,n,m]
  h16* qb = (h16*)(ws + o); o += (size_t)Bb * 2 * 512 * Nn * 2;  // q / probs
  float* sc = (float*)(ws + o); o += (size_t)Bb * 2 * 512 * KVv * 4; // scores/ctx
  float* st = (float*)(ws + o); o += 256;                        // {sum,ssq}x32
  if (ws_size < o) return;  // insufficient workspace
  h16* pb = qb;             // probs overlays q (q dead after scores)
  h16* cx = (h16*)sc;       // ctx overlays scores (scores dead after softmax)

  // ---- stage 1: K (transposed) and V projections, both heads ----
  {
    GP g{};
    g.A = emb_all; g.aS0 = 0;                 g.aS1 = (long)Nn * KVv;
    g.B = Wk;      g.bS0 = (long)KVv * KVv;   g.bS1 = 0;
    g.O = kf;      g.oS0 = (long)KVv * Nn;    g.oS1 = 2L * KVv * Nn;
    g.z0n = 2; g.nsrc = 1;
    g.lda = KVv; g.ldb = KVv; g.ldo = Nn; g.K = KVv;
    g.alpha = 1.f; g.flags = 1 | 2; g.stats = nullptr;
    gemm_wmma<float, float><<<dim3(KVv/64, Nn/64, Bb*2), 128, 0, stream>>>(g);
    g.B = Wv; g.O = vf; g.flags = 1;
    g.ldo = KVv; g.oS0 = (long)Nn * KVv; g.oS1 = 2L * Nn * KVv;
    gemm_wmma<float, float><<<dim3(KVv/64, Nn/64, Bb*2), 128, 0, stream>>>(g);
  }

  for (int i = 0; i < 4; ++i) {
    const int C = Cs[i];
    // ---- q[b,h,d,n] = Wq_h x emb^T ----
    {
      GP g{};
      g.A = Wq[i];  g.aS0 = (long)C * C;  g.aS1 = 0;
      g.B = emb[i]; g.bS0 = 0;            g.bS1 = (long)Nn * C;
      g.O = qb;     g.oS0 = (long)C * Nn; g.oS1 = 2L * C * Nn;
      g.z0n = 2; g.nsrc = 1;
      g.lda = C; g.ldb = C; g.ldo = Nn; g.K = C;
      g.alpha = 1.f; g.flags = 1; g.stats = nullptr;
      gemm_wmma<float, float><<<dim3(Nn/64, C/64, Bb*2), 128, 0, stream>>>(g);
    }
    zero_stats_kernel<<<1, 64, 0, stream>>>(st);
    // ---- scores[b,h,d,m] = SCALE * q x k, with fused mean/var stats ----
    {
      GP g{};
      g.A = qb; g.aS0 = (long)C * Nn;   g.aS1 = 2L * C * Nn;
      g.B = kf; g.bS0 = (long)KVv * Nn; g.bS1 = 2L * KVv * Nn;
      g.O = sc; g.oS0 = (long)C * KVv;  g.oS1 = 2L * C * KVv;
      g.z0n = 2; g.nsrc = 1;
      g.lda = Nn; g.ldb = Nn; g.ldo = KVv; g.K = Nn;
      g.alpha = SCALE; g.flags = 0; g.stats = st;
      gemm_wmma<h16, h16><<<dim3(KVv/64, C/64, Bb*2), 128, 0, stream>>>(g);
    }
    // ---- instance-norm + softmax -> probs f16 ----
    norm_softmax_kernel<<<dim3(C, Bb*2), 256, 0, stream>>>(sc, pb, st, C);
    // ---- ctx[b,n,d] = 0.5 * sum_h probs_h x v_h^T (transposed f16 out) ----
    {
      GP g{};
      g.A = pb; g.aS0 = 0; g.aS1 = 2L * C * KVv;  g.aSrc = (long)C * KVv;
      g.B = vf; g.bS0 = 0; g.bS1 = 2L * Nn * KVv; g.bSrc = (long)Nn * KVv;
      g.O = cx; g.oS0 = 0; g.oS1 = (long)Nn * C;
      g.z0n = 1; g.nsrc = 2;
      g.lda = KVv; g.ldb = KVv; g.ldo = C; g.K = KVv;
      g.alpha = 0.5f; g.flags = 1 | 2; g.stats = nullptr;
      gemm_wmma<h16, h16><<<dim3(Nn/64, C/64, Bb), 128, 0, stream>>>(g);
    }
    // ---- out slice[b,n,off+e] = ctx x Wo^T (fp32 into d_out) ----
    {
      GP g{};
      g.A = cx;    g.aS0 = 0; g.aS1 = (long)Nn * C;
      g.B = Wo[i]; g.bS0 = 0; g.bS1 = 0;
      g.O = (float*)d_out + off[i]; g.oS0 = 0; g.oS1 = (long)Nn * KVv;
      g.z0n = 1; g.nsrc = 1;
      g.lda = C; g.ldb = C; g.ldo = KVv; g.K = C;
      g.alpha = 1.f; g.flags = 0; g.stats = nullptr;
      gemm_wmma<h16, float><<<dim3(C/64, Nn/64, Bb), 128, 0, stream>>>(g);
    }
  }
}
